// SelfAttention_14877766713732
// MI455X (gfx1250) — matrix operand order repaired and checked by reading, not compiled, not run on hardware
//
#include <hip/hip_runtime.h>

// ---------------------------------------------------------------------------
// Self-attention forward for MI455X (gfx1250, wave32).
// v_wmma_f32_16x16x32_bf16 compute + double-buffered async global->LDS DMA.
// B=2, S=2048, D=1024, H=16, DK=64.
// ---------------------------------------------------------------------------

typedef __bf16 v16bf __attribute__((ext_vector_type(16)));
typedef __bf16 v8bf  __attribute__((ext_vector_type(8)));
typedef float  v8f   __attribute__((ext_vector_type(8)));

#define BATCH 2
#define SEQ   2048
#define DIM   1024
#define HEADS 16
#define DK    64
#define ROWS  (BATCH * SEQ)      // 4096

union BFrag { v16bf v; v8bf h[2]; };

static __device__ __forceinline__ v8f wmma_bf16(v16bf a, v16bf b, v8f c) {
  return __builtin_amdgcn_wmma_f32_16x16x32_bf16(false, a, false, b,
                                                 (short)0, c, false, false);
}

// Async global->LDS copy (GLOBAL_LOAD_ASYNC_TO_LDS_B128, ASYNCcnt-tracked).
// LDS operand = low 32 bits of the flat pointer.
static __device__ __forceinline__ void async_copy_b128(const void* g, void* l) {
  uint32_t lds_off = (uint32_t)(uintptr_t)l;
  asm volatile("global_load_async_to_lds_b128 %0, %1, off"
               :
               : "v"(lds_off), "v"(g)
               : "memory");
}
static __device__ __forceinline__ void wait_async() {
  asm volatile("s_wait_asynccnt 0x0" ::: "memory");
}

// ------------------------------ fp32 -> bf16 -------------------------------
__global__ __launch_bounds__(256) void cvt_f32_bf16_kernel(
    const float* __restrict__ src, __bf16* __restrict__ dst, int n) {
  int i = (blockIdx.x * blockDim.x + threadIdx.x) * 4;
  if (i + 3 < n) {
    float4 f = *(const float4*)(src + i);
    dst[i + 0] = (__bf16)f.x;
    dst[i + 1] = (__bf16)f.y;
    dst[i + 2] = (__bf16)f.z;
    dst[i + 3] = (__bf16)f.w;
  }
}

// ------------------------- Tiled WMMA GEMM (bf16) --------------------------
// C[M,N] = A[M,K] * B[K,N] + bias[N].
// Block tile 128x64, 8 waves, each wave computes 16 rows x 64 cols
// (4 accumulators of 16x16), K stepped by 32, double-buffered LDS.
// mode 0: scatter qkv -> Q[bh][s][dk], K[bh][s][dk], Vt[bh][dk][s] (bf16)
// mode 1: dense fp32 output (+bias)
#define TILE_M 128
#define TILE_N 64
#define TILE_K 32

__global__ __launch_bounds__(256) void gemm_bf16_kernel(
    const __bf16* __restrict__ A, const __bf16* __restrict__ B,
    const float* __restrict__ bias, int M, int K, int N, int mode,
    __bf16* __restrict__ q_out, __bf16* __restrict__ k_out,
    __bf16* __restrict__ vt_out, float* __restrict__ f_out) {
  __shared__ __bf16 ldsA[2][TILE_M * TILE_K];   // 2 x 8 KB, row-major [m][k]
  __shared__ __bf16 ldsBT[2][TILE_N * TILE_K];  // 2 x 4 KB, transposed [n][k]

  const int tid  = threadIdx.x;
  const int lane = tid & 31;
  const int wave = tid >> 5;
  const int hf   = lane >> 4;   // half-wave select
  const int l16  = lane & 15;
  const int m0   = blockIdx.x * TILE_M;
  const int n0   = blockIdx.y * TILE_N;
  const int waveRow = wave * 16;

  const int rA = tid >> 1;            // A staging: row 0..127
  const int cA = (tid & 1) * 16;      // A staging: col 0 or 16
  const int rB = tid >> 3;            // B staging: k row 0..31
  const int cB = (tid & 7) * 8;       // B staging: n col group

  auto stage = [&](int kk, int buf) {
    // A tile 128x32 via async DMA (2 x b128 per thread)
    const __bf16* srcA = A + (size_t)(m0 + rA) * K + kk + cA;
    __bf16* dstA = &ldsA[buf][rA * TILE_K + cA];
    async_copy_b128(srcA, dstA);
    async_copy_b128(srcA + 8, dstA + 8);
    // B tile 32x64 transposed -> [n][k] (needs reshape, regular path)
    uint4 d = *(const uint4*)(B + (size_t)(kk + rB) * N + n0 + cB);
    const __bf16* e = (const __bf16*)&d;
#pragma unroll
    for (int j = 0; j < 8; ++j) ldsBT[buf][(cB + j) * TILE_K + rB] = e[j];
  };

  v8f acc[4] = {};
  const int nTiles = K / TILE_K;

  stage(0, 0);
  for (int t = 0; t < nTiles; ++t) {
    wait_async();      // own async copies for tile t landed
    __syncthreads();   // => everyone's copies + ds stores visible
    if (t + 1 < nTiles) {
      stage((t + 1) * TILE_K, (t + 1) & 1);  // DMA overlaps compute below
      if (t + 2 < nTiles) {                  // warm L2 two k-steps ahead
        int k2 = (t + 2) * TILE_K;
        __builtin_prefetch(A + (size_t)(m0 + rA) * K + k2 + cA, 0, 1);
        __builtin_prefetch(B + (size_t)(k2 + rB) * N + n0 + cB, 0, 1);
      }
    }
    const __bf16* lA  = ldsA[t & 1];
    const __bf16* lBT = ldsBT[t & 1];

    // A fragment: row = l16; K chunks {hf*8 .. +7} and {16+hf*8 .. +7}
    BFrag af;
    const __bf16* arow = lA + (waveRow + l16) * TILE_K;
    af.h[0] = *(const v8bf*)(arow + hf * 8);
    af.h[1] = *(const v8bf*)(arow + 16 + hf * 8);

    // Preload all B fragments, then issue WMMAs back-to-back.
    BFrag bf[4];
#pragma unroll
    for (int nb = 0; nb < 4; ++nb) {
      const __bf16* bcol = lBT + (nb * 16 + l16) * TILE_K;
      bf[nb].h[0] = *(const v8bf*)(bcol + hf * 8);
      bf[nb].h[1] = *(const v8bf*)(bcol + 16 + hf * 8);
    }
#pragma unroll
    for (int nb = 0; nb < 4; ++nb)
      acc[nb] = wmma_bf16(af.v, bf[nb].v, acc[nb]);
  }

  // Epilogue: C layout row = r + 8*hf, col = l16 (per 16-wide n block).
#pragma unroll
  for (int nb = 0; nb < 4; ++nb) {
    int n_g = n0 + nb * 16 + l16;
    float bval = bias[n_g];
#pragma unroll
    for (int r = 0; r < 8; ++r) {
      int m_g = m0 + waveRow + r + 8 * hf;
      float v = acc[nb][r] + bval;
      if (mode == 1) {
        f_out[(size_t)m_g * N + n_g] = v;
      } else {
        int sec    = n_g >> 10;    // 0=Q 1=K 2=V
        int within = n_g & 1023;
        int hd     = within >> 6;  // head
        int dkk    = within & 63;
        int b      = m_g >> 11;
        int s      = m_g & 2047;
        int bh     = b * HEADS + hd;
        __bf16 hv = (__bf16)v;
        if (sec == 0)      q_out[((size_t)bh * SEQ + s) * DK + dkk] = hv;
        else if (sec == 1) k_out[((size_t)bh * SEQ + s) * DK + dkk] = hv;
        else               vt_out[((size_t)bh * DK + dkk) * SEQ + s] = hv;
      }
    }
  }
}

// ------------------------- Flash attention kernel --------------------------
// grid.x = B*H (32), grid.y = SEQ/128 (16). 8 waves, each wave owns 16 query
// rows for the whole key loop (no cross-wave softmax reductions).
// K/V tiles double-buffered and fetched with async DMA.
__global__ __launch_bounds__(256) void attn_kernel(
    const __bf16* __restrict__ Qb, const __bf16* __restrict__ Kb,
    const __bf16* __restrict__ Vt, __bf16* __restrict__ Ob) {
  __shared__ __bf16 ldsK[2][32 * DK];    // key tile  [k_local][d], 2 x 4 KB
  __shared__ __bf16 ldsV[2][DK * 32];    // V^T tile  [d][k_local], 2 x 4 KB
  __shared__ __bf16 ldsP[8 * 16 * 32];   // per-wave P tiles,       8 KB

  const int tid  = threadIdx.x;
  const int lane = tid & 31;
  const int wave = tid >> 5;
  const int hf   = lane >> 4;
  const int l16  = lane & 15;
  const int bh   = blockIdx.x;
  const int q0   = blockIdx.y * 128 + wave * 16;

  // Q fragments for d windows [0,32) and [32,64); A-layout, direct global.
  BFrag qf[2];
  const __bf16* qrow = Qb + ((size_t)bh * SEQ + q0 + l16) * DK;
#pragma unroll
  for (int dd = 0; dd < 2; ++dd) {
    qf[dd].h[0] = *(const v8bf*)(qrow + dd * 32 + hf * 8);
    qf[dd].h[1] = *(const v8bf*)(qrow + dd * 32 + 16 + hf * 8);
  }

  float mrow[8], lrow[8];
  v8f oacc[4] = {};
#pragma unroll
  for (int r = 0; r < 8; ++r) { mrow[r] = -1e30f; lrow[r] = 0.f; }

  const int rK = tid >> 3, cK = (tid & 7) * 8;  // K staging 32x64
  const int rV = tid >> 2, cV = (tid & 3) * 8;  // V staging 64x32

  auto stageKV = [&](int kt, int buf) {
    async_copy_b128(Kb + ((size_t)bh * SEQ + kt + rK) * DK + cK,
                    &ldsK[buf][rK * DK + cK]);
    async_copy_b128(Vt + ((size_t)bh * DK + rV) * SEQ + kt + cV,
                    &ldsV[buf][rV * 32 + cV]);
  };

  const int nTiles = SEQ / 32;
  stageKV(0, 0);
  for (int t = 0; t < nTiles; ++t) {
    wait_async();
    __syncthreads();
    if (t + 1 < nTiles) {
      stageKV((t + 1) * 32, (t + 1) & 1);  // DMA overlaps compute below
      if (t + 2 < nTiles) {
        int k2 = (t + 2) * 32;
        __builtin_prefetch(Kb + ((size_t)bh * SEQ + k2 + rK) * DK + cK, 0, 1);
        __builtin_prefetch(Vt + ((size_t)bh * DK + rV) * SEQ + k2 + cV, 0, 1);
      }
    }
    const __bf16* lK = ldsK[t & 1];
    const __bf16* lV = ldsV[t & 1];

    // scores S = Q * K^T for two 16-key blocks; preload all K fragments
    BFrag kf[2][2];
#pragma unroll
    for (int nb = 0; nb < 2; ++nb) {
#pragma unroll
      for (int dd = 0; dd < 2; ++dd) {
        const __bf16* kcol = lK + (nb * 16 + l16) * DK + dd * 32;
        kf[nb][dd].h[0] = *(const v8bf*)(kcol + hf * 8);
        kf[nb][dd].h[1] = *(const v8bf*)(kcol + 16 + hf * 8);
      }
    }
    v8f sc[2] = {};
#pragma unroll
    for (int nb = 0; nb < 2; ++nb)
#pragma unroll
      for (int dd = 0; dd < 2; ++dd)
        sc[nb] = wmma_bf16(qf[dd].v, kf[nb][dd].v, sc[nb]);

    // online softmax; row = r + 8*hf, cols live across the 16-lane half
#pragma unroll
    for (int r = 0; r < 8; ++r) {
      float s0 = sc[0][r] * 0.125f;  // 1/sqrt(64)
      float s1 = sc[1][r] * 0.125f;
      float vmax = fmaxf(s0, s1);
#pragma unroll
      for (int m = 1; m < 16; m <<= 1)
        vmax = fmaxf(vmax, __shfl_xor(vmax, m, 32));
      float mn = fmaxf(mrow[r], vmax);
      float alpha = __expf(mrow[r] - mn);
      float p0 = __expf(s0 - mn), p1 = __expf(s1 - mn);
      float psum = p0 + p1;
#pragma unroll
      for (int m = 1; m < 16; m <<= 1) psum += __shfl_xor(psum, m, 32);
      lrow[r] = lrow[r] * alpha + psum;
      mrow[r] = mn;
#pragma unroll
      for (int tt = 0; tt < 4; ++tt) oacc[tt][r] *= alpha;
      int row = r + 8 * hf;
      ldsP[(wave * 16 + row) * 32 + l16]      = (__bf16)p0;
      ldsP[(wave * 16 + row) * 32 + 16 + l16] = (__bf16)p1;
    }
    __syncthreads();  // P written in C-layout, re-read in A-layout

    // O += P (16x32) * V (32x64); preload all V fragments
    BFrag pf;
    const __bf16* prow = ldsP + (wave * 16 + l16) * 32;
    pf.h[0] = *(const v8bf*)(prow + hf * 8);
    pf.h[1] = *(const v8bf*)(prow + 16 + hf * 8);
    BFrag vf[4];
#pragma unroll
    for (int tt = 0; tt < 4; ++tt) {
      const __bf16* vcol = lV + (tt * 16 + l16) * 32;
      vf[tt].h[0] = *(const v8bf*)(vcol + hf * 8);
      vf[tt].h[1] = *(const v8bf*)(vcol + 16 + hf * 8);
    }
#pragma unroll
    for (int tt = 0; tt < 4; ++tt)
      oacc[tt] = wmma_bf16(pf.v, vf[tt].v, oacc[tt]);
  }

  // normalize and write O as [b][s][h*64+d] bf16 (row-major for out-proj GEMM)
  int b = bh >> 4, hd = bh & 15;
#pragma unroll
  for (int r = 0; r < 8; ++r) {
    float inv = 1.0f / lrow[r];
    int q = q0 + r + 8 * hf;
#pragma unroll
    for (int tt = 0; tt < 4; ++tt) {
      Ob[((size_t)b * SEQ + q) * DIM + hd * DK + tt * 16 + l16] =
          (__bf16)(oacc[tt][r] * inv);
    }
  }
}

// --------------------------------- launch ----------------------------------
extern "C" void kernel_launch(void* const* d_in, const int* in_sizes, int n_in,
                              void* d_out, int out_size, void* d_ws,
                              size_t ws_size, hipStream_t stream) {
  const float* x     = (const float*)d_in[0];  // [2,2048,1024]
  const float* W_qkv = (const float*)d_in[1];  // [1024,3072]
  const float* b_qkv = (const float*)d_in[2];  // [3072]
  const float* W_out = (const float*)d_in[3];  // [1024,1024]
  const float* b_out = (const float*)d_in[4];  // [1024]
  float* out = (float*)d_out;                  // [2,2048,1024] fp32

  const size_t nX    = (size_t)ROWS * DIM;        // 4,194,304
  const size_t nWqkv = (size_t)DIM * 3 * DIM;     // 3,145,728
  const size_t nWout = (size_t)DIM * DIM;         // 1,048,576
  const size_t nHead = (size_t)BATCH * HEADS * SEQ * DK;  // 4,194,304

  char* ws = (char*)d_ws;
  __bf16* xb    = (__bf16*)ws; ws += nX * 2;
  __bf16* wqkvb = (__bf16*)ws; ws += nWqkv * 2;
  __bf16* woutb = (__bf16*)ws; ws += nWout * 2;
  __bf16* qb    = (__bf16*)ws; ws += nHead * 2;
  __bf16* kb    = (__bf16*)ws; ws += nHead * 2;
  __bf16* vt    = (__bf16*)ws; ws += nHead * 2;
  __bf16* ob    = (__bf16*)ws; ws += nX * 2;     // ~48 MB total

  // 1) fp32 -> bf16
  cvt_f32_bf16_kernel<<<nX / 1024, 256, 0, stream>>>(x, xb, (int)nX);
  cvt_f32_bf16_kernel<<<nWqkv / 1024, 256, 0, stream>>>(W_qkv, wqkvb, (int)nWqkv);
  cvt_f32_bf16_kernel<<<nWout / 1024, 256, 0, stream>>>(W_out, woutb, (int)nWout);

  // 2) QKV projection: [4096,1024] x [1024,3072] + bias -> Q/K/Vt scatter
  gemm_bf16_kernel<<<dim3(ROWS / TILE_M, (3 * DIM) / TILE_N), 256, 0, stream>>>(
      xb, wqkvb, b_qkv, ROWS, DIM, 3 * DIM, /*mode=*/0, qb, kb, vt, nullptr);

  // 3) flash attention per (b,h) x q-tile
  attn_kernel<<<dim3(BATCH * HEADS, SEQ / 128), 256, 0, stream>>>(qb, kb, vt, ob);

  // 4) output projection: [4096,1024] x [1024,1024] + bias -> fp32 out
  gemm_bf16_kernel<<<dim3(ROWS / TILE_M, DIM / TILE_N), 256, 0, stream>>>(
      ob, woutb, b_out, ROWS, DIM, DIM, /*mode=*/1, nullptr, nullptr, nullptr,
      out);
}
